// CrossModalCluesMining_22488448762722
// MI455X (gfx1250) — matrix operand order repaired and checked by reading, not compiled, not run on hardware
//
#include <hip/hip_runtime.h>

// CrossModalCluesMining for MI455X (gfx1250, wave32, WMMA, TDM).
// Pipeline: f32->bf16 convert, WMMA QKV GEMMs, flash attention
// (wmma 16x16x32 bf16, online softmax, double-buffered TDM-staged K/V tiles),
// WMMA out-proj, residual+LayerNorm, WMMA enhancement GEMM, scalar tail.
// GEMM/attention inner loops are software-pipelined: the Tensor Data Mover
// fills LDS buffer (i+1) while WMMAs consume buffer (i); one barrier/iter.

#define D_   256
#define NH_  8
#define HD_  32
#define B_   16
#define S_   1024
#define M_   (B_ * S_)      // 16384 rows
#define EPS_ 1e-5f

typedef __bf16 bf16;
typedef __attribute__((ext_vector_type(16))) __bf16 v16bf;
typedef __attribute__((ext_vector_type(8)))  float  v8f;
typedef __attribute__((ext_vector_type(4)))  unsigned int u32x4;
typedef __attribute__((ext_vector_type(8)))  int          i32x8;
typedef __attribute__((ext_vector_type(4)))  int          i32x4;

#if __has_builtin(__builtin_amdgcn_tensor_load_to_lds)
#define USE_TDM 1
#else
#define USE_TDM 0
#endif

// ---------------------------------------------------------------- TDM helpers
// Issue a 2D tile load Global->LDS through the Tensor Data Mover.
// tile_w/tile_h in elements (2B each), stride = tensor row stride in elements.
__device__ __forceinline__ void tdm_load_2d(const void* gsrc, void* ldst,
                                            unsigned tile_w, unsigned tile_h,
                                            unsigned stride) {
#if USE_TDM
  const unsigned long long ga = (unsigned long long)(uintptr_t)gsrc;
  const unsigned ldsoff = (unsigned)(uintptr_t)ldst;  // low 32 bits = LDS offset
  u32x4 g0 = {0u, 0u, 0u, 0u};
  g0[0] = 1u;                                  // count=1, user mode
  g0[1] = ldsoff;                              // lds_addr (bytes)
  g0[2] = (unsigned)ga;                        // global_addr[31:0]
  g0[3] = (unsigned)(ga >> 32) | (2u << 30);   // global_addr[56:32] | type=2
  i32x8 g1 = {0, 0, 0, 0, 0, 0, 0, 0};
  const unsigned td0 = stride;                 // tensor width >= tile (no OOB)
  const unsigned td1 = 0x7fffffffu;            // plenty of rows
  g1[0] = (int)(1u << 16);                                   // data_size=1 (2B)
  g1[1] = (int)((td0 & 0xffffu) << 16);                      // tensor_dim0[15:0]
  g1[2] = (int)((td0 >> 16) | ((td1 & 0xffffu) << 16));      // td0 hi | td1 lo
  g1[3] = (int)((td1 >> 16) | (tile_w << 16));               // td1 hi | tile_dim0
  g1[4] = (int)(tile_h & 0xffffu);                           // tile_dim1 (dim2=0)
  g1[5] = (int)stride;                                       // dim0_stride[31:0]
  g1[6] = 0;                                                 // dim0_stride[47:32]
  g1[7] = 0;
  i32x4 z4 = {0, 0, 0, 0};
#if __clang_major__ >= 23
  i32x8 z8 = {0, 0, 0, 0, 0, 0, 0, 0};
  __builtin_amdgcn_tensor_load_to_lds(g0, g1, z4, z4, z8, 0);
#else
  __builtin_amdgcn_tensor_load_to_lds(g0, g1, z4, z4, 0);
#endif
#else
  (void)gsrc; (void)ldst; (void)tile_w; (void)tile_h; (void)stride;
#endif
}

__device__ __forceinline__ void tdm_wait() {
#if USE_TDM
#if __has_builtin(__builtin_amdgcn_s_wait_tensorcnt)
  __builtin_amdgcn_s_wait_tensorcnt(0);
#else
  asm volatile("s_wait_tensorcnt 0" ::: "memory");
#endif
#endif
}

// ---------------------------------------------------------------- WMMA helpers
__device__ __forceinline__ v8f wmma_bf16(v16bf a, v16bf b, v8f c) {
  return __builtin_amdgcn_wmma_f32_16x16x32_bf16(
      false, a, false, b, (short)0, c, false, false);
}

// A fragment (16x32) from row-major [16][32] LDS tile (one 32B run per lane).
__device__ __forceinline__ v16bf load_afrag(const bf16* tile, int lane) {
  const int lo = lane & 15, hi = lane >> 4;
  v16bf a;
#pragma unroll
  for (int v = 0; v < 8; ++v) {
    const int kb = hi * 16 + 2 * v;
    a[2 * v]     = tile[lo * 32 + kb];
    a[2 * v + 1] = tile[lo * 32 + kb + 1];
  }
  return a;
}

// B fragment (32x16) from an n-major tile (tile[n*32 + k]); one 32B run per lane.
__device__ __forceinline__ v16bf load_bfrag_nmajor(const bf16* tile, int lane) {
  const int lo = lane & 15, hi = lane >> 4;
  v16bf b;
#pragma unroll
  for (int v = 0; v < 8; ++v) {
    const int k0 = hi * 16 + 2 * v;
    b[2 * v]     = tile[lo * 32 + k0];
    b[2 * v + 1] = tile[lo * 32 + k0 + 1];
  }
  return b;
}

// B fragment (32x16) from a k-major tile [k][n], row stride 32 elements.
__device__ __forceinline__ v16bf load_bfrag_kmajor(const bf16* tile, int lane) {
  const int lo = lane & 15, hi = lane >> 4;
  v16bf b;
#pragma unroll
  for (int v = 0; v < 8; ++v) {
    const int k0 = hi * 16 + 2 * v;
    b[2 * v]     = tile[k0 * 32 + lo];
    b[2 * v + 1] = tile[(k0 + 1) * 32 + lo];
  }
  return b;
}

// ---------------------------------------------------------------- convert f32 -> bf16
__global__ void cvt_kernel(const float* __restrict__ src, bf16* __restrict__ dst, int n) {
  int i = blockIdx.x * blockDim.x + threadIdx.x;
  if (i < n) dst[i] = (bf16)src[i];
}

// ---------------------------------------------------------------- generic WMMA GEMM
// C[M_,256] = A[M_,K](bf16) @ W[K,256](bf16) + bias. Block: 32x64 tile, 4 waves.
// Double-buffered LDS; TDM fills A tile (i+1) while wmmas consume tile (i).
template <typename OutT, int K>
__global__ __launch_bounds__(128) void gemm_bf16_wmma(
    const bf16* __restrict__ A, const bf16* __restrict__ W,
    const float* __restrict__ bias, OutT* __restrict__ out) {
  constexpr int N = 256;
  constexpr int NIT = K / 32;
  __shared__ __align__(16) bf16 sA[2][32 * 32];       // [m][k] tile (TDM target)
  __shared__ __align__(16) bf16 sBT[2][4 * 16 * 32];  // per-wave n-major B tile

  const int tid = threadIdx.x;
  const int w = tid >> 5, lane = tid & 31;
  const int lo = lane & 15, hi = lane >> 4;
  const int m0 = blockIdx.x * 32;
  const int n0 = blockIdx.y * 64 + w * 16;

  auto stage = [&](int i) {
    const int kk = i * 32;
    bf16* a_dst = sA[i & 1];
    bf16* b_dst = sBT[i & 1] + w * 16 * 32;
#if USE_TDM
    if (w == 0) tdm_load_2d(A + (size_t)m0 * K + kk, a_dst, 32, 32, K);
#else
    {  // stage A tile [32][32]: one 16B chunk per thread, coalesced
      const int e = tid * 8, m = e >> 5, kl = e & 31;
      *(uint4*)(a_dst + e) = *(const uint4*)(A + (size_t)(m0 + m) * K + kk + kl);
    }
#endif
    {  // stage B tile transposed: lane = local k row, then n-major scatter
      const bf16* wsrc = W + (size_t)(kk + lane) * N + n0;
      if (kk + 32 < K) __builtin_prefetch(wsrc + 32 * N, 0, 1);
      bf16 hbuf[16];
      *(uint4*)(hbuf)     = *(const uint4*)(wsrc);
      *(uint4*)(hbuf + 8) = *(const uint4*)(wsrc + 8);
#pragma unroll
      for (int j = 0; j < 16; ++j) b_dst[j * 32 + lane] = hbuf[j];
    }
  };

  v8f acc0 = {}, acc1 = {};
  stage(0);
  for (int i = 0; i < NIT; ++i) {
#if USE_TDM
    if (w == 0) tdm_wait();     // tile i landed (only tile i outstanding here)
#endif
    __syncthreads();            // stage(i) visible; readers of buf (i+1)&1 done
    if (i + 1 < NIT) stage(i + 1);  // overlaps with compute below
    const bf16* a_t = sA[i & 1];
    const bf16* b_t = sBT[i & 1] + w * 16 * 32;
    const v16bf bfrag = load_bfrag_nmajor(b_t, lane);
    const v16bf af0 = load_afrag(a_t, lane);
    const v16bf af1 = load_afrag(a_t + 16 * 32, lane);
    acc0 = wmma_bf16(af0, bfrag, acc0);
    acc1 = wmma_bf16(af1, bfrag, acc1);
  }

  const int n = n0 + lo;
  const float bv = bias[n];
#pragma unroll
  for (int r = 0; r < 8; ++r) {
    out[(size_t)(m0 + r + 8 * hi) * N + n]      = (OutT)(acc0[r] + bv);
    out[(size_t)(m0 + 16 + r + 8 * hi) * N + n] = (OutT)(acc1[r] + bv);
  }
}

// ---------------------------------------------------------------- flash attention
// 4 waves / block, wave w owns q-tile (blockIdx.x*4 + w). K/V tiles (32 keys)
// double-buffered: TDM fills (jj+1) while all waves compute on (jj).
__global__ __launch_bounds__(128) void flash_attn_kernel(
    const bf16* __restrict__ Q, const bf16* __restrict__ K,
    const bf16* __restrict__ V, bf16* __restrict__ O) {
  __shared__ __align__(16) bf16 qs[4 * 16 * 32];     // per-wave [m][dim], scaled
  __shared__ __align__(16) bf16 ks[2][32 * 32];      // [key][dim]
  __shared__ __align__(16) bf16 vs[2][32 * 32];      // [key][dim]
  __shared__ __align__(16) bf16 ps[4 * 16 * 32];     // per-wave [m][key]

  const int tid = threadIdx.x;
  const int w = tid >> 5, lane = tid & 31;
  const int lo = lane & 15, hi = lane >> 4;
  const int h = blockIdx.y, b = blockIdx.z;
  const int qrow0 = (blockIdx.x * 4 + w) * 16;
  const float scale = 0.1767766952966369f;  // 1/sqrt(32)

  bf16* qtile = qs + w * 512;
  bf16* ptile = ps + w * 512;

  auto stageKV = [&](int jj) {
    bf16* kd = ks[jj & 1];
    bf16* vd = vs[jj & 1];
    const bf16* kbase = K + (size_t)(b * S_ + jj * 32) * D_ + h * HD_;
    const bf16* vbase = V + (size_t)(b * S_ + jj * 32) * D_ + h * HD_;
#if USE_TDM
    if (w == 0) {
      tdm_load_2d(kbase, kd, 32, 32, D_);
      tdm_load_2d(vbase, vd, 32, 32, D_);
    }
#else
    {  // cooperative stage: 32 rows x 64B, one 16B chunk per thread
      const int row = tid >> 2, part = tid & 3;
      *(uint4*)(kd + row * 32 + part * 8) =
          *(const uint4*)(kbase + (size_t)row * D_ + part * 8);
      *(uint4*)(vd + row * 32 + part * 8) =
          *(const uint4*)(vbase + (size_t)row * D_ + part * 8);
    }
#endif
  };

  {  // each wave stages its own scaled Q tile (same-wave DS ordering suffices)
    const int m = lane >> 1, koff = (lane & 1) * 16;
    const bf16* src = Q + (size_t)(b * S_ + qrow0 + m) * D_ + h * HD_ + koff;
#pragma unroll
    for (int j = 0; j < 16; ++j) qtile[m * 32 + koff + j] = (bf16)((float)src[j] * scale);
  }
  const v16bf qf = load_afrag(qtile, lane);

  float mrow[8], lrow[8];
  v8f o0 = {}, o1 = {};
#pragma unroll
  for (int r = 0; r < 8; ++r) { mrow[r] = -3.0e38f; lrow[r] = 0.0f; }

  stageKV(0);
  for (int jj = 0; jj < S_ / 32; ++jj) {
#if USE_TDM
    if (w == 0) tdm_wait();     // K/V of tile jj landed
#endif
    __syncthreads();            // stage(jj) visible; readers of alt buffer done
    if (jj + 1 < S_ / 32) stageKV(jj + 1);  // overlaps with compute below
    const bf16* kt = ks[jj & 1];
    const bf16* vt = vs[jj & 1];

    const v8f z = {};
    v8f s0 = wmma_bf16(qf, load_bfrag_nmajor(kt, lane), z);            // keys +0..15
    v8f s1 = wmma_bf16(qf, load_bfrag_nmajor(kt + 16 * 32, lane), z);  // keys +16..31

    // online softmax: row m's 16 columns live across one 16-lane half
#pragma unroll
    for (int r = 0; r < 8; ++r) {
      float mx = fmaxf(s0[r], s1[r]);
#pragma unroll
      for (int off = 8; off > 0; off >>= 1) mx = fmaxf(mx, __shfl_xor(mx, off, 16));
      const float mnew  = fmaxf(mrow[r], mx);
      const float alpha = __expf(mrow[r] - mnew);
      const float p0 = __expf(s0[r] - mnew);
      const float p1 = __expf(s1[r] - mnew);
      float rs = p0 + p1;
#pragma unroll
      for (int off = 8; off > 0; off >>= 1) rs += __shfl_xor(rs, off, 16);
      lrow[r] = lrow[r] * alpha + rs;
      mrow[r] = mnew;
      o0[r] *= alpha;
      o1[r] *= alpha;
      ptile[(r + 8 * hi) * 32 + lo]      = (bf16)p0;  // C-frag -> A-frag via LDS
      ptile[(r + 8 * hi) * 32 + 16 + lo] = (bf16)p1;
    }
    const v16bf pf = load_afrag(ptile, lane);
    o0 = wmma_bf16(pf, load_bfrag_kmajor(vt, lane), o0);       // dims 0-15
    o1 = wmma_bf16(pf, load_bfrag_kmajor(vt + 16, lane), o1);  // dims 16-31
  }

  bf16* dst = O + (size_t)(b * S_ + qrow0) * D_ + h * HD_;
#pragma unroll
  for (int r = 0; r < 8; ++r) {
    const int m = r + 8 * hi;
    const float inv = 1.0f / lrow[r];
    dst[(size_t)m * D_ + lo]      = (bf16)(o0[r] * inv);
    dst[(size_t)m * D_ + 16 + lo] = (bf16)(o1[r] * inv);
  }
}

// ---------------------------------------------------------------- residual + LayerNorm
__global__ __launch_bounds__(256) void ln_residual_kernel(
    const float* __restrict__ x, const float* __restrict__ resid,
    const float* __restrict__ g, const float* __restrict__ bb,
    bf16* __restrict__ comb, int col0) {
  const int wid = threadIdx.x >> 5, lane = threadIdx.x & 31;
  const int row = blockIdx.x * 8 + wid;
  float v[8], s = 0.0f, sq = 0.0f;
#pragma unroll
  for (int i = 0; i < 8; ++i) {
    const int c = i * 32 + lane;
    const float t = x[(size_t)row * D_ + c] + resid[(size_t)row * D_ + c];
    v[i] = t; s += t; sq += t * t;
  }
#pragma unroll
  for (int off = 16; off > 0; off >>= 1) {
    s  += __shfl_xor(s, off, 32);
    sq += __shfl_xor(sq, off, 32);
  }
  const float mu   = s * (1.0f / D_);
  const float var  = sq * (1.0f / D_) - mu * mu;
  const float rstd = rsqrtf(var + EPS_);
#pragma unroll
  for (int i = 0; i < 8; ++i) {
    const int c = i * 32 + lane;
    comb[(size_t)row * (2 * D_) + col0 + c] = (bf16)((v[i] - mu) * rstd * g[c] + bb[c]);
  }
}

// ---------------------------------------------------------------- inconsistency + tails
__global__ __launch_bounds__(256) void inc_tail_kernel(
    const float* __restrict__ is, const float* __restrict__ ts,
    const float* __restrict__ incw, const float* __restrict__ incb,
    float* __restrict__ out) {
  const int idx = blockIdx.x * blockDim.x + threadIdx.x;  // 0..4095
  const int bb = idx >> 8, n = idx & 255;
  float s = incb[n];
#pragma unroll 4
  for (int k = 0; k < 256; ++k) {
    const float a = is[bb * 256 + k], t = ts[bb * 256 + k];
    s += (a - t) * incw[(size_t)k * 256 + n];
    s += (a * t) * incw[(size_t)(256 + k) * 256 + n];
  }
  out[4194304 + idx]        = s;        // inconsistency
  out[4194304 + 4096 + idx] = is[idx];  // img_specific passthrough
  out[4194304 + 8192 + idx] = ts[idx];  // txt_specific passthrough
}

// ================================================================ host launcher
extern "C" void kernel_launch(void* const* d_in, const int* in_sizes, int n_in,
                              void* d_out, int out_size, void* d_ws, size_t ws_size,
                              hipStream_t stream) {
  (void)in_sizes; (void)n_in; (void)out_size; (void)ws_size;
  const float* img_c = (const float*)d_in[0];
  const float* txt_c = (const float*)d_in[1];
  const float* img_s = (const float*)d_in[2];
  const float* txt_s = (const float*)d_in[3];
  const float* i2t_wq = (const float*)d_in[4];   const float* i2t_bq = (const float*)d_in[5];
  const float* i2t_wk = (const float*)d_in[6];   const float* i2t_bk = (const float*)d_in[7];
  const float* i2t_wv = (const float*)d_in[8];   const float* i2t_bv = (const float*)d_in[9];
  const float* i2t_wo = (const float*)d_in[10];  const float* i2t_bo = (const float*)d_in[11];
  const float* t2i_wq = (const float*)d_in[12];  const float* t2i_bq = (const float*)d_in[13];
  const float* t2i_wk = (const float*)d_in[14];  const float* t2i_bk = (const float*)d_in[15];
  const float* t2i_wv = (const float*)d_in[16];  const float* t2i_bv = (const float*)d_in[17];
  const float* t2i_wo = (const float*)d_in[18];  const float* t2i_bo = (const float*)d_in[19];
  const float* ln_ig  = (const float*)d_in[20];  const float* ln_ib  = (const float*)d_in[21];
  const float* ln_tg  = (const float*)d_in[22];  const float* ln_tb  = (const float*)d_in[23];
  const float* enh_w  = (const float*)d_in[24];  const float* enh_b  = (const float*)d_in[25];
  const float* inc_w  = (const float*)d_in[26];  const float* inc_b  = (const float*)d_in[27];

  char* ws = (char*)d_ws;
  size_t off = 0;
  auto alloc = [&](size_t bytes) -> void* {
    void* p = ws + off;
    off += (bytes + 255) & ~(size_t)255;
    return p;
  };
  const size_t MD2 = (size_t)M_ * D_ * 2;
  bf16* imgc_bf = (bf16*)alloc(MD2);
  bf16* txtc_bf = (bf16*)alloc(MD2);
  bf16* wbf[8];
  for (int i = 0; i < 8; ++i) wbf[i] = (bf16*)alloc((size_t)D_ * D_ * 2);
  bf16* enhw_bf = (bf16*)alloc((size_t)2 * D_ * D_ * 2);
  bf16* Qi = (bf16*)alloc(MD2);  bf16* Ki = (bf16*)alloc(MD2);  bf16* Vi = (bf16*)alloc(MD2);
  bf16* Qt = (bf16*)alloc(MD2);  bf16* Kt = (bf16*)alloc(MD2);  bf16* Vt = (bf16*)alloc(MD2);
  bf16* AOi = (bf16*)alloc(MD2);
  bf16* AOt = (bf16*)alloc(MD2);
  float* Pi = (float*)alloc((size_t)M_ * D_ * 4);
  float* Pt = (float*)alloc((size_t)M_ * D_ * 4);
  bf16* comb = (bf16*)alloc((size_t)M_ * 2 * D_ * 2);

  const int CT = 256;
  const int nMD = M_ * D_, nW = D_ * D_, nEW = 2 * D_ * D_;
  cvt_kernel<<<(nMD + CT - 1) / CT, CT, 0, stream>>>(img_c, imgc_bf, nMD);
  cvt_kernel<<<(nMD + CT - 1) / CT, CT, 0, stream>>>(txt_c, txtc_bf, nMD);
  const float* wsrc[8] = {i2t_wq, i2t_wk, i2t_wv, i2t_wo, t2i_wq, t2i_wk, t2i_wv, t2i_wo};
  for (int i = 0; i < 8; ++i)
    cvt_kernel<<<(nW + CT - 1) / CT, CT, 0, stream>>>(wsrc[i], wbf[i], nW);
  cvt_kernel<<<(nEW + CT - 1) / CT, CT, 0, stream>>>(enh_w, enhw_bf, nEW);

  dim3 gg(M_ / 32, D_ / 64);  // 512 x 4, 128 threads
  // i2t: Q from img_common; K,V from txt_common
  gemm_bf16_wmma<bf16, 256><<<gg, 128, 0, stream>>>(imgc_bf, wbf[0], i2t_bq, Qi);
  gemm_bf16_wmma<bf16, 256><<<gg, 128, 0, stream>>>(txtc_bf, wbf[1], i2t_bk, Ki);
  gemm_bf16_wmma<bf16, 256><<<gg, 128, 0, stream>>>(txtc_bf, wbf[2], i2t_bv, Vi);
  // t2i: Q from txt_common; K,V from img_common
  gemm_bf16_wmma<bf16, 256><<<gg, 128, 0, stream>>>(txtc_bf, wbf[4], t2i_bq, Qt);
  gemm_bf16_wmma<bf16, 256><<<gg, 128, 0, stream>>>(imgc_bf, wbf[5], t2i_bk, Kt);
  gemm_bf16_wmma<bf16, 256><<<gg, 128, 0, stream>>>(imgc_bf, wbf[6], t2i_bv, Vt);

  dim3 ag(S_ / 64, NH_, B_);  // 16 x 8 x 16, 128 threads
  flash_attn_kernel<<<ag, 128, 0, stream>>>(Qi, Ki, Vi, AOi);
  flash_attn_kernel<<<ag, 128, 0, stream>>>(Qt, Kt, Vt, AOt);

  gemm_bf16_wmma<float, 256><<<gg, 128, 0, stream>>>(AOi, wbf[3], i2t_bo, Pi);
  gemm_bf16_wmma<float, 256><<<gg, 128, 0, stream>>>(AOt, wbf[7], t2i_bo, Pt);

  ln_residual_kernel<<<M_ / 8, 256, 0, stream>>>(Pi, img_c, ln_ig, ln_ib, comb, 0);
  ln_residual_kernel<<<M_ / 8, 256, 0, stream>>>(Pt, txt_c, ln_tg, ln_tb, comb, D_);

  // enhancement: [M, 2D] @ [2D, D] + b -> d_out[0 : M*D] (f32)
  gemm_bf16_wmma<float, 512><<<gg, 128, 0, stream>>>(comb, enhw_bf, enh_b, (float*)d_out);

  inc_tail_kernel<<<16, 256, 0, stream>>>(img_s, txt_s, inc_w, inc_b, (float*)d_out);
}
